// STAN_88545045775036
// MI455X (gfx1250) — matrix operand order repaired and checked by reading, not compiled, not run on hardware
//
#include <hip/hip_runtime.h>

typedef _Float16 v16h __attribute__((ext_vector_type(16)));
typedef _Float16 v8h  __attribute__((ext_vector_type(8)));
typedef float    v8f  __attribute__((ext_vector_type(8)));

#define DEV __device__ __forceinline__

constexpr int B    = 8;
constexpr int S    = 256;
constexpr int D    = 128;
constexpr int H    = 8;
constexpr int HD   = 16;
constexpr int NL   = 2;
constexpr int NPOI = 50000;
constexpr int BS   = B * S;              // 2048 rows of activations
constexpr float LN_EPS      = 1e-5f;
constexpr float INV_SQRT_HD = 0.25f;     // 1/sqrt(16)
constexpr float DEG2RAD     = 0.017453292519943295f;

// ---------------------------------------------------------------------------
// Helpers
// ---------------------------------------------------------------------------
DEV v16h concat8(v8h lo, v8h hi) {
  return __builtin_shufflevector(lo, hi, 0,1,2,3,4,5,6,7,8,9,10,11,12,13,14,15);
}

// ---------------------------------------------------------------------------
// Weight cast + transpose: wT[l][n][k] = (f16) w[l][k][n]  (WMMA B-operand
// loads along K become a single contiguous 32-byte v16h load per lane)
// ---------------------------------------------------------------------------
__global__ void castw_kernel(const float* __restrict__ w, _Float16* __restrict__ wT) {
  int idx = blockIdx.x * blockDim.x + threadIdx.x;
  if (idx >= NL * D * D) return;
  int l = idx / (D * D);
  int rem = idx % (D * D);
  int k = rem / D, n = rem % D;
  wT[l * D * D + n * D + k] = (_Float16)w[idx];
}

// ---------------------------------------------------------------------------
// Spatio-temporal bias scalars: ab[l][h] = (e_dt . wst[l,:,h], e_ds . wst[l,:,h])
// ---------------------------------------------------------------------------
__global__ void ab_kernel(const float* __restrict__ e_dt, const float* __restrict__ e_ds,
                          const float* __restrict__ wst, float* __restrict__ ab) {
  int t = threadIdx.x;
  if (t >= NL * H) return;
  int l = t / H, h = t % H;
  float a = 0.f, bb = 0.f;
  for (int d = 0; d < D; ++d) {
    float w = wst[(l * D + d) * H + h];
    a  += e_dt[d] * w;
    bb += e_ds[d] * w;
  }
  ab[t * 2 + 0] = a;
  ab[t * 2 + 1] = bb;
}

// ---------------------------------------------------------------------------
// Input embeddings: x = poi_emb[poi_ids] + user_emb[user_id]  (f32 + f16 copy)
// ---------------------------------------------------------------------------
__global__ void embed_kernel(const int* __restrict__ poi_ids, const int* __restrict__ user_ids,
                             const float* __restrict__ poi_emb, const float* __restrict__ user_emb,
                             float* __restrict__ x, _Float16* __restrict__ xh) {
  int bs = blockIdx.x;
  int d  = threadIdx.x;
  int b  = bs / S;
  int pid = poi_ids[bs];
  int uid = user_ids[b];
  float v = poi_emb[pid * D + d] + user_emb[uid * D + d];
  x [bs * D + d] = v;
  xh[bs * D + d] = (_Float16)v;
}

// ---------------------------------------------------------------------------
// dt (hours, clamp 500) and haversine ds (km/100, clamp 1000), [B,S,S]
// ---------------------------------------------------------------------------
__global__ void dtds_kernel(const float* __restrict__ ts, const float* __restrict__ gps,
                            float* __restrict__ dt, float* __restrict__ dsb) {
  int idx = blockIdx.x * blockDim.x + threadIdx.x;
  if (idx >= B * S * S) return;
  int j = idx % S;
  int i = (idx / S) % S;
  int b = idx / (S * S);
  float ti = ts[b * S + i], tj = ts[b * S + j];
  float dtv = fminf(fabsf(ti - tj) * (1.0f / 3600.0f), 500.0f);
  float lati = gps[(b * S + i) * 2 + 0] * DEG2RAD;
  float loni = gps[(b * S + i) * 2 + 1] * DEG2RAD;
  float latj = gps[(b * S + j) * 2 + 0] * DEG2RAD;
  float lonj = gps[(b * S + j) * 2 + 1] * DEG2RAD;
  float sdlat = sinf((lati - latj) * 0.5f);
  float sdlon = sinf((loni - lonj) * 0.5f);
  float a = sdlat * sdlat + cosf(lati) * cosf(latj) * sdlon * sdlon;
  a = fminf(fmaxf(a, 0.0f), 1.0f);
  float c = 2.0f * asinf(sqrtf(a));
  float dsv = fminf(6371.0f * c * 0.01f, 1000.0f);
  dt [idx] = dtv;
  dsb[idx] = dsv;
}

// ---------------------------------------------------------------------------
// WMMA GEMM, 16x64 strip per wave: out[M,D] = A[M,D](f16) @ W + bias
// A tile loaded once per K-step, reused across 4 N-tiles (4 accumulators).
// MODE 0: write f16 row-major                       (Q projection)
// MODE 1: write f32 row-major                       (output projection)
// MODE 2: write f16 transposed  vt[b][h][hd][S]     (V projection)
// MODE 3: write f16 zero-padded khp[b][s][h][32]    (K projection; pads K=32
//         tile so the scores kernel loads it with no lane predication)
// ---------------------------------------------------------------------------
template <int MODE>
__global__ void gemm_d128(const _Float16* __restrict__ A, const _Float16* __restrict__ Bt,
                          const float* __restrict__ bias, void* __restrict__ out, int M) {
  const int lane = threadIdx.x & 31;
  const int wid  = threadIdx.x >> 5;
  const int tile = blockIdx.x * 4 + wid;          // (M/16) * 2 strips
  if (tile >= (M / 16) * 2) return;
  const int mt    = tile >> 1;
  const int ncol0 = (tile & 1) * 64;

  const int m   = lane & 15;
  const int kb  = (lane >> 4) * 8;
  const int n   = lane & 15;
  const int kb2 = (lane >> 4) * 16;
  const int row = mt * 16 + m;

  v8f acc[4] = {{}, {}, {}, {}};
#pragma unroll
  for (int kt = 0; kt < D / 32; ++kt) {
    const int ks = kt * 32;
    v16h av = *(const v16h*)(A + row * D + ks + kb2);
#pragma unroll
    for (int t = 0; t < 4; ++t) {
      const int col = ncol0 + t * 16 + n;
      v16h bv = *(const v16h*)(Bt + col * D + ks + kb2);
      acc[t] = __builtin_amdgcn_wmma_f32_16x16x32_f16(false, av, false, bv, (short)0,
                                                      acc[t], false, false);
    }
  }
  const int mb = (lane >> 4) * 8;
#pragma unroll
  for (int t = 0; t < 4; ++t) {
    const int col = ncol0 + t * 16 + n;
    const float bc = bias[col];
#pragma unroll
    for (int r = 0; r < 8; ++r) {
      const int rr = mt * 16 + mb + r;
      const float v = acc[t][r] + bc;
      if constexpr (MODE == 0) {
        ((_Float16*)out)[rr * D + col] = (_Float16)v;
      } else if constexpr (MODE == 1) {
        ((float*)out)[rr * D + col] = v;
      } else if constexpr (MODE == 2) {            // V transposed: vt[b][h][hd][S]
        const int b = rr >> 8;                     // rr / S
        const int s = rr & (S - 1);
        const int hh = col >> 4;
        const int nn = col & 15;
        ((_Float16*)out)[((b * H + hh) * HD + nn) * S + s] = (_Float16)v;
      } else {                                     // K padded: khp[bs][h][32]
        const int hh = col >> 4;
        const int nn = col & 15;
        _Float16* kp = (_Float16*)out + (rr * H + hh) * 32;
        kp[nn]      = (_Float16)v;
        kp[nn + 16] = (_Float16)0.f;               // zero pad for K=16..31
      }
    }
  }
}

// ---------------------------------------------------------------------------
// Raw attention scores: one wave per (b,h,i-tile), covering 4 j-tiles;
// Q tile (K=hd=16 zero-padded to 32) loaded once, 4 WMMAs.
// K comes from the zero-padded khp layout, so every lane issues two
// unconditional 16B loads, at kb and 16+kb (no EXEC predication).
// ---------------------------------------------------------------------------
__global__ void scores_kernel(const _Float16* __restrict__ qh, const _Float16* __restrict__ khp,
                              float* __restrict__ scores) {
  const int lane = threadIdx.x & 31;
  const int wid  = threadIdx.x >> 5;
  const int bh = blockIdx.x;
  const int b  = bh >> 3, h = bh & 7;
  const int it = blockIdx.y;
  const int jt0 = wid * 4;                         // 4 waves x 4 j-tiles = 16

  const int m   = lane & 15;
  const int kb  = (lane >> 4) * 8;                 // 0 or 8 (< HD); K 16..31 zero
  const int n   = lane & 15;
  const int kb2 = (lane >> 4) * 16;
  const int i   = it * 16 + m;

  v8h z = {};
  v8h a0 = *(const v8h*)(qh + (b * S + i) * D + h * HD + kb);
  v16h av = concat8(a0, z);

  v8f acc[4] = {{}, {}, {}, {}};
#pragma unroll
  for (int t = 0; t < 4; ++t) {
    const int j = (jt0 + t) * 16 + n;
    v16h bv = concat8(*(const v8h*)(khp + ((b * S + j) * H + h) * 32 + kb),
                      *(const v8h*)(khp + ((b * S + j) * H + h) * 32 + 16 + kb));
    acc[t] = __builtin_amdgcn_wmma_f32_16x16x32_f16(false, av, false, bv, (short)0,
                                                    acc[t], false, false);
  }
  const int mb = (lane >> 4) * 8;
#pragma unroll
  for (int t = 0; t < 4; ++t) {
    const int j = (jt0 + t) * 16 + n;
#pragma unroll
    for (int r = 0; r < 8; ++r)
      scores[(bh * S + it * 16 + mb + r) * S + j] = acc[t][r];
  }
}

// ---------------------------------------------------------------------------
// Scale + st-bias + key-padding mask + softmax -> attn (f16).
// Wave32 shuffle reductions, 2 barriers per row.
// ---------------------------------------------------------------------------
__global__ void softmax_kernel(const float* __restrict__ scores, const float* __restrict__ dt,
                               const float* __restrict__ dsb, const int* __restrict__ lengths,
                               const float* __restrict__ ab, int layer,
                               _Float16* __restrict__ attnh) {
  __shared__ float redmax[8];
  __shared__ float redsum[8];
  const int r  = blockIdx.x;                       // (b*H+h)*S + i
  const int i  = r % S;
  const int bh = r / S;
  const int b  = bh >> 3, h = bh & 7;
  const int j  = threadIdx.x;
  const int lane = j & 31, wid = j >> 5;

  const float a  = ab[(layer * H + h) * 2 + 0];
  const float bb = ab[(layer * H + h) * 2 + 1];
  const int len  = lengths[b];

  float v = scores[r * S + j] * INV_SQRT_HD
          + a  * dt [(b * S + i) * S + j]
          + bb * dsb[(b * S + i) * S + j];
  if (j >= len) v = -10000.0f;

  float wm = v;
#pragma unroll
  for (int off = 16; off; off >>= 1) wm = fmaxf(wm, __shfl_xor(wm, off, 32));
  if (lane == 0) redmax[wid] = wm;
  __syncthreads();
  float mx = redmax[0];
#pragma unroll
  for (int w = 1; w < 8; ++w) mx = fmaxf(mx, redmax[w]);

  float e = expf(v - mx);
  float ws = e;
#pragma unroll
  for (int off = 16; off; off >>= 1) ws += __shfl_xor(ws, off, 32);
  if (lane == 0) redsum[wid] = ws;
  __syncthreads();
  float total = 0.f;
#pragma unroll
  for (int w = 0; w < 8; ++w) total += redsum[w];

  attnh[r * S + j] = (_Float16)(e / total);
}

// ---------------------------------------------------------------------------
// out[b,i,h,:] = attn[b,h,i,:] @ V  (V pre-transposed: vt[b][h][hd][S], so the
// B operand is one contiguous 32B v16h load per K-step)
// ---------------------------------------------------------------------------
__global__ void av_kernel(const _Float16* __restrict__ attnh, const _Float16* __restrict__ vt,
                          _Float16* __restrict__ avh) {
  const int lane = threadIdx.x & 31;
  const int wid  = threadIdx.x >> 5;
  const int bh = blockIdx.x;
  const int b  = bh >> 3, h = bh & 7;
  const int it = blockIdx.y * 4 + wid;

  const int m   = lane & 15;
  const int kb  = (lane >> 4) * 8;
  const int n   = lane & 15;
  const int kb2 = (lane >> 4) * 16;
  const int i   = it * 16 + m;

  v8f c = {};
#pragma unroll
  for (int kt = 0; kt < S / 32; ++kt) {
    const int ks = kt * 32;
    v16h av = *(const v16h*)(attnh + (bh * S + i) * S + ks + kb2);
    v16h bv = *(const v16h*)(vt + ((bh)*HD + n) * S + ks + kb2);
    c = __builtin_amdgcn_wmma_f32_16x16x32_f16(false, av, false, bv, (short)0, c,
                                               false, false);
  }
  const int mb = (lane >> 4) * 8;
#pragma unroll
  for (int r = 0; r < 8; ++r)
    avh[(b * S + it * 16 + mb + r) * D + h * HD + n] = (_Float16)c[r];
}

// ---------------------------------------------------------------------------
// Residual + LayerNorm; updates x (f32) and xh (f16) in place.
// ---------------------------------------------------------------------------
__global__ void ln_kernel(const float* __restrict__ proj, const float* __restrict__ ln_g,
                          const float* __restrict__ ln_b, int layer,
                          float* __restrict__ x, _Float16* __restrict__ xh) {
  __shared__ float red[D];
  const int row = blockIdx.x;
  const int d   = threadIdx.x;
  float r = x[row * D + d] + proj[row * D + d];

  red[d] = r; __syncthreads();
  for (int off = D / 2; off; off >>= 1) {
    if (d < off) red[d] += red[d + off];
    __syncthreads();
  }
  float mu = red[0] * (1.0f / D); __syncthreads();

  float dv = r - mu;
  red[d] = dv * dv; __syncthreads();
  for (int off = D / 2; off; off >>= 1) {
    if (d < off) red[d] += red[d + off];
    __syncthreads();
  }
  float var = red[0] * (1.0f / D);
  float y = ln_g[layer * D + d] * dv * rsqrtf(var + LN_EPS) + ln_b[layer * D + d];
  x [row * D + d] = y;
  xh[row * D + d] = (_Float16)y;
}

// ---------------------------------------------------------------------------
// History sum over valid positions: xs[b,d] = sum_{s<len[b]} x[b,s,d]
// ---------------------------------------------------------------------------
__global__ void xs_kernel(const float* __restrict__ x, const int* __restrict__ lengths,
                          float* __restrict__ xs) {
  int b = blockIdx.x, d = threadIdx.x;
  int len = lengths[b];
  float s = 0.f;
  for (int t = 0; t < len; ++t) s += x[(b * S + t) * D + d];
  xs[b * D + d] = s;
}

// ---------------------------------------------------------------------------
// Matching: out[b,l] = (poi_emb[l] . xs[b]) / valid[b]; one wave per poi row.
// ---------------------------------------------------------------------------
__global__ void match_kernel(const float* __restrict__ poi_emb, const float* __restrict__ xs,
                             const int* __restrict__ lengths, float* __restrict__ out) {
  __shared__ float sxs[B * D];
  for (int idx = threadIdx.x; idx < B * D; idx += blockDim.x) sxs[idx] = xs[idx];
  __syncthreads();

  const int lane = threadIdx.x & 31;
  const int wid  = threadIdx.x >> 5;
  const int l = blockIdx.x * 8 + wid;
  if (l >= NPOI) return;

  float4 pv = *(const float4*)(poi_emb + l * D + lane * 4);
  float acc[B];
#pragma unroll
  for (int b = 0; b < B; ++b) {
    const float* xb = sxs + b * D + lane * 4;
    acc[b] = pv.x * xb[0] + pv.y * xb[1] + pv.z * xb[2] + pv.w * xb[3];
  }
#pragma unroll
  for (int off = 16; off; off >>= 1) {
#pragma unroll
    for (int b = 0; b < B; ++b) acc[b] += __shfl_down(acc[b], off, 32);
  }
  if (lane == 0) {
#pragma unroll
    for (int b = 0; b < B; ++b) {
      float pad = (float)(S - lengths[b]);
      if (pad < 1.0f) pad = 1.0f;                  // reference: max(pad_count, 1)
      out[b * NPOI + l] = acc[b] / ((float)S - pad);
    }
  }
}

// ---------------------------------------------------------------------------
// Host launcher
// ---------------------------------------------------------------------------
extern "C" void kernel_launch(void* const* d_in, const int* in_sizes, int n_in,
                              void* d_out, int out_size, void* d_ws, size_t ws_size,
                              hipStream_t stream) {
  const int*   poi_ids  = (const int*)  d_in[0];
  const int*   user_ids = (const int*)  d_in[1];
  const float* ts       = (const float*)d_in[2];
  const float* gps      = (const float*)d_in[3];
  const int*   lengths  = (const int*)  d_in[4];
  const float* poi_emb  = (const float*)d_in[5];
  const float* user_emb = (const float*)d_in[6];
  const float* e_dt     = (const float*)d_in[7];
  const float* e_ds     = (const float*)d_in[8];
  const float* wq       = (const float*)d_in[9];
  const float* p_bq     = (const float*)d_in[10];
  const float* wk       = (const float*)d_in[11];
  const float* p_bk     = (const float*)d_in[12];
  const float* wv       = (const float*)d_in[13];
  const float* p_bv     = (const float*)d_in[14];
  const float* wo       = (const float*)d_in[15];
  const float* p_bo     = (const float*)d_in[16];
  const float* wst      = (const float*)d_in[17];
  const float* ln_g     = (const float*)d_in[18];
  const float* ln_b     = (const float*)d_in[19];
  float* out = (float*)d_out;

  char* p = (char*)d_ws;
  auto take = [&](size_t bytes) -> char* {
    char* r = p;
    p += (bytes + 255) & ~(size_t)255;
    return r;
  };
  float*    x      = (float*)   take(sizeof(float)    * BS * D);
  float*    proj   = (float*)   take(sizeof(float)    * BS * D);
  float*    scores = (float*)   take(sizeof(float)    * (size_t)B * H * S * S);
  float*    dt     = (float*)   take(sizeof(float)    * (size_t)B * S * S);
  float*    dsb    = (float*)   take(sizeof(float)    * (size_t)B * S * S);
  float*    ab     = (float*)   take(sizeof(float)    * NL * H * 2);
  float*    xs     = (float*)   take(sizeof(float)    * B * D);
  _Float16* xh     = (_Float16*)take(sizeof(_Float16) * BS * D);
  _Float16* qh     = (_Float16*)take(sizeof(_Float16) * BS * D);
  _Float16* khp    = (_Float16*)take(sizeof(_Float16) * BS * H * 32); // zero-padded K
  _Float16* vt     = (_Float16*)take(sizeof(_Float16) * BS * D);      // [b][h][hd][S]
  _Float16* avh    = (_Float16*)take(sizeof(_Float16) * BS * D);
  _Float16* attnh  = (_Float16*)take(sizeof(_Float16) * (size_t)B * H * S * S);
  _Float16* wqT    = (_Float16*)take(sizeof(_Float16) * NL * D * D);
  _Float16* wkT    = (_Float16*)take(sizeof(_Float16) * NL * D * D);
  _Float16* wvT    = (_Float16*)take(sizeof(_Float16) * NL * D * D);
  _Float16* woT    = (_Float16*)take(sizeof(_Float16) * NL * D * D);

  const int castBlocks = (NL * D * D + 255) / 256;
  castw_kernel<<<castBlocks, 256, 0, stream>>>(wq, wqT);
  castw_kernel<<<castBlocks, 256, 0, stream>>>(wk, wkT);
  castw_kernel<<<castBlocks, 256, 0, stream>>>(wv, wvT);
  castw_kernel<<<castBlocks, 256, 0, stream>>>(wo, woT);
  ab_kernel<<<1, 32, 0, stream>>>(e_dt, e_ds, wst, ab);
  embed_kernel<<<BS, D, 0, stream>>>(poi_ids, user_ids, poi_emb, user_emb, x, xh);
  dtds_kernel<<<(B * S * S + 255) / 256, 256, 0, stream>>>(ts, gps, dt, dsb);

  const int gemmBlocks = (BS / 16) * 2 / 4;        // 256 strip-tiles, 4 waves per block
  for (int l = 0; l < NL; ++l) {
    const _Float16* WQ = wqT + (size_t)l * D * D;
    const _Float16* WK = wkT + (size_t)l * D * D;
    const _Float16* WV = wvT + (size_t)l * D * D;
    const _Float16* WO = woT + (size_t)l * D * D;

    gemm_d128<0><<<gemmBlocks, 128, 0, stream>>>(xh, WQ, p_bq + l * D, qh, BS);
    gemm_d128<3><<<gemmBlocks, 128, 0, stream>>>(xh, WK, p_bk + l * D, khp, BS);
    gemm_d128<2><<<gemmBlocks, 128, 0, stream>>>(xh, WV, p_bv + l * D, vt, BS);

    scores_kernel<<<dim3(B * H, S / 16), 128, 0, stream>>>(qh, khp, scores);
    softmax_kernel<<<B * H * S, S, 0, stream>>>(scores, dt, dsb, lengths, ab, l, attnh);
    av_kernel<<<dim3(B * H, S / 16 / 4), 128, 0, stream>>>(attnh, vt, avh);

    gemm_d128<1><<<gemmBlocks, 128, 0, stream>>>(avh, WO, p_bo + l * D, proj, BS);
    ln_kernel<<<BS, D, 0, stream>>>(proj, ln_g, ln_b, l, x, xh);
  }

  xs_kernel<<<B, D, 0, stream>>>(x, lengths, xs);
  match_kernel<<<(NPOI + 7) / 8, 256, 0, stream>>>(poi_emb, xs, lengths, out);
}
